// DynamicPointConvBasicBlock_71184787964123
// MI455X (gfx1250) — matrix operand order repaired and checked by reading, not compiled, not run on hardware
//
#include <hip/hip_runtime.h>
#include <hip/hip_bf16.h>

typedef __attribute__((ext_vector_type(16))) _Float16 v16h;
typedef __attribute__((ext_vector_type(8)))  _Float16 v8h;
typedef __attribute__((ext_vector_type(8)))  float    v8f;

#define NPTS   200000
#define CIN    32
#define COUT   64
#define K3     27
#define NT32   (NPTS/32)          // 6250 supertiles of 32 rows (exact)
#define LN_EPS 1e-3f

// d_out layout (floats), reference tuple concatenated flat:
#define OFF_COORS 0                               // N*3
#define OFF_OUT   600000                          // N*64
#define OFF_NUM   13400000                        // B=4
#define OFF_VIDX  13400004                        // N*27
#define OFF_CIDX  18800004                        // N
#define WPACK_ELEMS (K3*4*32*16)                  // 55296 halfs = 110592 B

// ---------- pre-pass: features f32 -> f16 ----------
__global__ void feat_to_half(const float* __restrict__ f, _Float16* __restrict__ h, int n) {
    for (int i = blockIdx.x * blockDim.x + threadIdx.x; i < n; i += gridDim.x * blockDim.x)
        h[i] = (_Float16)f[i];
}

// ---------- pre-pass: pack W into WMMA B-fragment order ----------
// wp[((t*4+nt)*32 + lane)*16 + kk] = W[col = nt*16 + (lane&15)][k = t*32 + (kk<8 ? 8*(lane>>4)+kk : 16+8*(lane>>4)+kk-8)]
__global__ void pack_w(const float* __restrict__ W, _Float16* __restrict__ wp) {
    int e = blockIdx.x * blockDim.x + threadIdx.x;
    if (e >= WPACK_ELEMS) return;
    int kk   = e & 15;
    int lane = (e >> 4) & 31;
    int nt   = (e >> 9) & 3;
    int t    = e >> 11;
    int col  = nt * 16 + (lane & 15);
    int k    = t * 32 + ((kk & 8) << 1) + ((lane >> 4) << 3) + (kk & 7);
    wp[e] = (_Float16)W[col * (K3 * CIN) + k];
}

// ---------- pass-through tuple outputs ----------
__global__ void aux_out(const float* __restrict__ coors,
                        const int* __restrict__ numl,
                        const int* __restrict__ vox,
                        const int* __restrict__ cid,
                        float* __restrict__ out) {
    const int total = 600000 + 4 + NPTS * K3 + NPTS;
    for (int i = blockIdx.x * blockDim.x + threadIdx.x; i < total; i += gridDim.x * blockDim.x) {
        if (i < 600000) {
            out[OFF_COORS + i] = coors[i];
        } else if (i < 600004) {
            out[OFF_NUM + (i - 600000)] = (float)numl[i - 600000];
        } else if (i < 600004 + NPTS * K3) {
            int j = i - 600004;
            int row = j / K3;
            int t   = j - row * K3;
            out[OFF_VIDX + j] = (float)vox[cid[row] * K3 + t];
        } else {
            int j = i - (600004 + NPTS * K3);
            out[OFF_CIDX + j] = (float)cid[j];
        }
    }
}

// Build a 16x32 f16 A fragment (documented gfx1250 VGPR layout) for one tap.
//   lane<16  : K {0..7} then {16..23} of row M=lane
//   lane>=16 : K {8..15} then {24..31} of row M=lane-16
__device__ __forceinline__ v16h load_a_frag(const _Float16* __restrict__ featH,
                                            int idx, int h) {
    int sidx = idx < 0 ? 0 : idx;
    const _Float16* src = featH + (size_t)sidx * CIN + h * 8;
    v8h lo = *(const v8h*)(src);
    v8h hi = *(const v8h*)(src + 16);
    if (idx < 0) {
        const v8h zh = {};
        lo = zh; hi = zh;
    }
    v16h a;
#pragma unroll
    for (int i = 0; i < 8; i++) { a[i] = lo[i]; a[i + 8] = hi[i]; }
    return a;
}

// ---------- fused gather + GEMM(WMMA f16) + LayerNorm + ReLU ----------
// Each wave owns a 32-row supertile (two 16-row WMMA tiles) so every B
// fragment fetched from LDS feeds TWO v_wmma ops.
__global__ void __launch_bounds__(256)
dpconv_wmma(const _Float16* __restrict__ featH,
            const _Float16* __restrict__ wpack,
            const int* __restrict__ vox,
            const int* __restrict__ cid,
            const float* __restrict__ gamma,
            const float* __restrict__ beta,
            float* __restrict__ outp) {
    extern __shared__ _Float16 s_w[];   // WPACK_ELEMS halfs = 110592 bytes

    // cooperative LDS fill of packed W (128-bit chunks)
    {
        const uint4* src = (const uint4*)wpack;
        uint4* dst = (uint4*)s_w;
        const int n16 = WPACK_ELEMS / 8;   // 6912 x 16B
        for (int i = threadIdx.x; i < n16; i += blockDim.x) dst[i] = src[i];
    }
    __syncthreads();

    const int lane = threadIdx.x & 31;
    const int wave = threadIdx.x >> 5;
    const int m    = lane & 15;   // row (for A) / column (for B,C) within tile
    const int h    = lane >> 4;   // lane half selector

    float g[4], bt[4];
#pragma unroll
    for (int nt = 0; nt < 4; nt++) {
        int col = nt * 16 + m;
        g[nt]  = gamma[col];
        bt[nt] = beta[col];
    }

    const int waveGlobal = blockIdx.x * 8 + wave;
    const int waveStride = gridDim.x * 8;

    for (int st = waveGlobal; st < NT32; st += waveStride) {
        const int row0 = st * 32 + m;          // tile 0 rows
        const int row1 = row0 + 16;            // tile 1 rows
        const int* vrow0 = vox + (size_t)cid[row0] * K3;
        const int* vrow1 = vox + (size_t)cid[row1] * K3;

        v8f acc[2][4];
        const v8f zacc = {};
#pragma unroll
        for (int p = 0; p < 2; p++)
#pragma unroll
            for (int nt = 0; nt < 4; nt++) acc[p][nt] = zacc;

        // manual pipeline: indices for tap t+1 fetched (and their feature
        // rows prefetched) while tap t's WMMAs run
        int i0 = vrow0[0];
        int i1 = vrow1[0];

        for (int t = 0; t < K3; t++) {
            int n0 = 0, n1 = 0;
            if (t + 1 < K3) {
                n0 = vrow0[t + 1];
                n1 = vrow1[t + 1];
                __builtin_prefetch(featH + (size_t)(n0 < 0 ? 0 : n0) * CIN, 0, 3);
                __builtin_prefetch(featH + (size_t)(n1 < 0 ? 0 : n1) * CIN, 0, 3);
            }

            v16h a0 = load_a_frag(featH, i0, h);
            v16h a1 = load_a_frag(featH, i1, h);

            const _Float16* wb = s_w + t * 2048 + lane * 16;
#pragma unroll
            for (int nt = 0; nt < 4; nt++) {
                v16h b = *(const v16h*)(wb + nt * 512);
                acc[0][nt] = __builtin_amdgcn_wmma_f32_16x16x32_f16(
                    false, a0, false, b, (short)0, acc[0][nt], false, false);
                acc[1][nt] = __builtin_amdgcn_wmma_f32_16x16x32_f16(
                    false, a1, false, b, (short)0, acc[1][nt], false, false);
            }
            i0 = n0;
            i1 = n1;
        }

        // Fused LayerNorm over 64 cols per row (rows live at C-VGPR v, half h),
        // then ReLU + store. Same epilogue for both 16-row tiles.
#pragma unroll
        for (int p = 0; p < 2; p++) {
            float mu[8], rs[8];
#pragma unroll
            for (int v = 0; v < 8; v++) {
                float s1 = acc[p][0][v] + acc[p][1][v] + acc[p][2][v] + acc[p][3][v];
                float s2 = acc[p][0][v] * acc[p][0][v] + acc[p][1][v] * acc[p][1][v] +
                           acc[p][2][v] * acc[p][2][v] + acc[p][3][v] * acc[p][3][v];
#pragma unroll
                for (int mask = 1; mask < 16; mask <<= 1) {
                    s1 += __shfl_xor(s1, mask, 32);
                    s2 += __shfl_xor(s2, mask, 32);
                }
                float mean = s1 * (1.f / COUT);
                float var  = s2 * (1.f / COUT) - mean * mean;
                mu[v] = mean;
                rs[v] = rsqrtf(var + LN_EPS);
            }
#pragma unroll
            for (int v = 0; v < 8; v++) {
                const int r = st * 32 + p * 16 + v + 8 * h;
                float* po = outp + (size_t)r * COUT + m;
#pragma unroll
                for (int nt = 0; nt < 4; nt++) {
                    float val = (acc[p][nt][v] - mu[v]) * rs[v] * g[nt] + bt[nt];
                    po[nt * 16] = val > 0.f ? val : 0.f;
                }
            }
        }
    }
}

extern "C" void kernel_launch(void* const* d_in, const int* in_sizes, int n_in,
                              void* d_out, int out_size, void* d_ws, size_t ws_size,
                              hipStream_t stream) {
    (void)in_sizes; (void)n_in; (void)out_size; (void)ws_size;
    const float* coors = (const float*)d_in[0];
    const float* feats = (const float*)d_in[1];
    const int*   numl  = (const int*)d_in[2];
    const int*   vox   = (const int*)d_in[3];
    const int*   cidx  = (const int*)d_in[4];
    // d_in[5] = mem_saving (unused)
    const float* W     = (const float*)d_in[6];
    const float* gamma = (const float*)d_in[7];
    const float* beta  = (const float*)d_in[8];
    float* out = (float*)d_out;

    _Float16* featH = (_Float16*)d_ws;                                   // 12.8 MB
    _Float16* wpack = (_Float16*)((char*)d_ws + (size_t)NPTS * CIN * 2); // +110.6 KB

    feat_to_half<<<2048, 256, 0, stream>>>(feats, featH, NPTS * CIN);
    pack_w<<<(WPACK_ELEMS + 255) / 256, 256, 0, stream>>>(W, wpack);
    aux_out<<<4096, 256, 0, stream>>>(coors, numl, vox, cidx, out);
    dpconv_wmma<<<512, 256, WPACK_ELEMS * sizeof(_Float16), stream>>>(
        featH, wpack, vox, cidx, gamma, beta, out + OFF_OUT);
}
